// GptOssMoE_39084202393885
// MI455X (gfx1250) — matrix operand order repaired and checked by reading, not compiled, not run on hardware
//
#include <hip/hip_runtime.h>
#include <cstdint>
#include <cstddef>

// Problem constants (reference: T=2048, D=1024, F=1024, E=16, top_k=4)
#define DDIM 1024
#define FDIM 1024
#define NEXP 16
#define TOPK 4
#define SWIGLU_LIMIT 7.0f
#define SWIGLU_ALPHA 1.702f

typedef __attribute__((ext_vector_type(16))) __bf16 v16bf;
typedef __attribute__((ext_vector_type(8)))  float v8f;

union FragBf { unsigned int u[8]; v16bf v; };

// fp32 -> bf16, round-to-nearest (ties away): 1 VALU op per element
__device__ __forceinline__ unsigned short f2bf(float f) {
  union { float f; unsigned int u; } c; c.f = f;
  return (unsigned short)((c.u + 0x8000u) >> 16);
}
// pack two fp32 -> bf16x2: two v_add + one v_perm_b32
__device__ __forceinline__ unsigned int pk2(float lo, float hi) {
  union { float f; unsigned int u; } a, b; a.f = lo; b.f = hi;
  unsigned int al = a.u + 0x8000u;
  unsigned int bh = b.u + 0x8000u;
  // dst bytes {0,1} = al bytes {2,3}; dst bytes {2,3} = bh bytes {2,3}
  return __builtin_amdgcn_perm(bh, al, 0x07060302u);
}

// generic (flat) pointer to LDS -> raw LDS byte offset (low 32 bits)
__device__ __forceinline__ unsigned int lds_addr(const void* p) {
  return (unsigned int)(uintptr_t)p;
}

// ---------------------------------------------------------------- control ---
__global__ void zero_ctrl_kernel(int* counts, int* cursors) {
  int i = threadIdx.x;
  if (i < NEXP) { counts[i] = 0; cursors[i] = 0; }
}

__global__ void zero_out_kernel(float* __restrict__ out, int n) {
  int i = blockIdx.x * blockDim.x + threadIdx.x;
  if (i < n) out[i] = 0.0f;
}

// ---------------------------------------------------------------- router ----
__global__ void router_kernel(const float* __restrict__ x,
                              const float* __restrict__ rw,
                              const float* __restrict__ rb,
                              int*   __restrict__ counts,
                              int*   __restrict__ topk_idx,
                              float* __restrict__ topk_wt,
                              int T) {
  int t = blockIdx.x * blockDim.x + threadIdx.x;
  if (t >= T) return;
  float acc[NEXP];
  #pragma unroll
  for (int e = 0; e < NEXP; ++e) acc[e] = rb[e];
  const float* xr = x + (size_t)t * DDIM;
  for (int d = 0; d < DDIM; ++d) {
    float xv = xr[d];
    const float* wr = rw + (size_t)d * NEXP;
    #pragma unroll
    for (int e = 0; e < NEXP; ++e) acc[e] = fmaf(xv, wr[e], acc[e]);
  }
  unsigned chosen = 0u;
  float vals[TOPK]; int idxs[TOPK];
  #pragma unroll
  for (int k = 0; k < TOPK; ++k) {
    float best = -3.4e38f; int bi = 0;
    #pragma unroll
    for (int e = 0; e < NEXP; ++e) {
      bool fr = !((chosen >> e) & 1u);
      if (fr && acc[e] > best) { best = acc[e]; bi = e; }
    }
    chosen |= (1u << bi); vals[k] = best; idxs[k] = bi;
  }
  float m = vals[0], s = 0.0f, w[TOPK];
  #pragma unroll
  for (int k = 0; k < TOPK; ++k) { w[k] = __expf(vals[k] - m); s += w[k]; }
  float inv = 1.0f / s;
  #pragma unroll
  for (int k = 0; k < TOPK; ++k) {
    topk_idx[t * TOPK + k] = idxs[k];
    topk_wt [t * TOPK + k] = w[k] * inv;
    atomicAdd(&counts[idxs[k]], 1);
  }
}

__global__ void scan_kernel(const int* __restrict__ counts,
                            int* __restrict__ offsets,
                            int* __restrict__ cursors) {
  if (threadIdx.x == 0) {
    int off = 0;
    for (int e = 0; e < NEXP; ++e) { offsets[e] = off; off += counts[e]; cursors[e] = 0; }
  }
}

__global__ void scatter_kernel(const int*   __restrict__ topk_idx,
                               const float* __restrict__ topk_wt,
                               const int*   __restrict__ offsets,
                               int*   __restrict__ cursors,
                               int*   __restrict__ tok_list,
                               float* __restrict__ wt_list,
                               int T) {
  int t = blockIdx.x * blockDim.x + threadIdx.x;
  if (t >= T) return;
  #pragma unroll
  for (int k = 0; k < TOPK; ++k) {
    int e = topk_idx[t * TOPK + k];
    int p = atomicAdd(&cursors[e], 1);
    int g = offsets[e] + p;
    tok_list[g] = t;
    wt_list [g] = topk_wt[t * TOPK + k];
  }
}

// ----------------------------------------------------------- GEMM kernels ---
// Block: 256 threads = 8 wave32. Tile: M=128, N=128, K-step=32, double-buffered.
// LDS A: row-major bf16, padded stride -> A frag = 2x ds_load_b128.
// LDS B: fragment-contiguous dwords -> B frag = 2x ds_load_b128, no repacking.
#define ASTRIDE 40   // LDS A row stride in bf16 elems (row = 80 B, 16B aligned)

// mlp1: act[slot, f] = swiglu( x[tok]*W1_gate , x[tok]*W1_up ), f-tile of 64.
__global__ void __launch_bounds__(256, 1)
moe_mlp1_kernel(const float* __restrict__ x,
                const float* __restrict__ w1,
                const int*   __restrict__ counts,
                const int*   __restrict__ offsets,
                const int*   __restrict__ tok_list,
                unsigned short* __restrict__ act) {
  const int e     = blockIdx.z;
  const int mtile = blockIdx.y;
  const int f0    = blockIdx.x * 64;
  const int rows  = counts[e];
  if (mtile * 128 >= rows) return;
  const int slotBase = offsets[e];

  __shared__ unsigned short ldsA[2][128 * ASTRIDE];
  __shared__ unsigned int   ldsB[2][2048];

  const int tid  = threadIdx.x;
  const int wave = tid >> 5;
  const int lane = tid & 31;
  const int h    = lane >> 4;
  const int ln   = lane & 15;

  // A fill: 128 rows x 32 cols, 16 elems/thread
  const int arow = tid >> 1;
  const int acol = (tid & 1) * 16;
  const int gRowA = mtile * 128 + arow;
  const bool aValid = gRowA < rows;
  const int tokA = aValid ? tok_list[slotBase + gRowA] : 0;
  const float* aSrc = x + (size_t)tokA * DDIM + acol;

  // B fill: thread owns k-pair (2 rows) x 8 n-cols
  const int kp = tid >> 4;
  const int nb = (tid & 15) * 8;
  const int bh = kp >> 3;
  const int bp = kp & 7;
  const int w1col = (nb < 64) ? (f0 + nb) : (FDIM + f0 + (nb - 64));
  const size_t ldb = 2 * FDIM;
  const float* bSrc0 = w1 + (size_t)e * DDIM * ldb + (size_t)(2 * kp) * ldb + w1col;

  v8f acc[8];
  #pragma unroll
  for (int j = 0; j < 8; ++j) acc[j] = (v8f){0,0,0,0,0,0,0,0};

  float4 aS0, aS1, aS2, aS3;
  float4 bL0, bL1, bH0, bH1;

  auto loadA = [&](int kt) {
    if (aValid) {
      const float4* s = (const float4*)(aSrc + kt * 32);
      aS0 = s[0]; aS1 = s[1]; aS2 = s[2]; aS3 = s[3];
    }
  };
  auto storeA = [&](int buf) {
    unsigned int* d = (unsigned int*)&ldsA[buf][arow * ASTRIDE + acol];
    if (aValid) {
      d[0] = pk2(aS0.x, aS0.y); d[1] = pk2(aS0.z, aS0.w);
      d[2] = pk2(aS1.x, aS1.y); d[3] = pk2(aS1.z, aS1.w);
      d[4] = pk2(aS2.x, aS2.y); d[5] = pk2(aS2.z, aS2.w);
      d[6] = pk2(aS3.x, aS3.y); d[7] = pk2(aS3.z, aS3.w);
    } else {
      #pragma unroll
      for (int i = 0; i < 8; ++i) d[i] = 0u;
    }
  };
  auto loadB = [&](int kt) {
    const float* r0 = bSrc0 + (size_t)kt * 32 * ldb;
    const float* r1 = r0 + ldb;
    bL0 = ((const float4*)r0)[0]; bL1 = ((const float4*)r0)[1];
    bH0 = ((const float4*)r1)[0]; bH1 = ((const float4*)r1)[1];
  };
  auto storeB = [&](int buf) {
    unsigned int* d = &ldsB[buf][0];
    const float lo[8] = {bL0.x,bL0.y,bL0.z,bL0.w, bL1.x,bL1.y,bL1.z,bL1.w};
    const float hi[8] = {bH0.x,bH0.y,bH0.z,bH0.w, bH1.x,bH1.y,bH1.z,bH1.w};
    #pragma unroll
    for (int i = 0; i < 8; ++i)
      d[((nb + i) * 2 + bh) * 8 + bp] = pk2(lo[i], hi[i]);
  };
  auto compute = [&](int buf) {
    FragBf a;
    #pragma unroll
    for (int v = 0; v < 8; ++v) {
      int kk = 16 * h + 2 * v;
      a.u[v] = *(const unsigned int*)&ldsA[buf][(wave * 16 + ln) * ASTRIDE + kk];
    }
    #pragma unroll
    for (int j = 0; j < 8; ++j) {
      FragBf b;
      const unsigned int* bpt = &ldsB[buf][((j * 16 + ln) * 2 + h) * 8];
      #pragma unroll
      for (int v = 0; v < 8; ++v) b.u[v] = bpt[v];
      acc[j] = __builtin_amdgcn_wmma_f32_16x16x32_bf16(
                 false, a.v, false, b.v, (short)0, acc[j], false, false);
    }
  };

  const int NK = DDIM / 32;
  loadA(0); loadB(0); storeA(0); storeB(0);
  __syncthreads();
  for (int kt = 0; kt < NK; ++kt) {
    const int cur = kt & 1;
    if (kt + 1 < NK) { loadA(kt + 1); loadB(kt + 1); }
    if (kt + 2 < NK)
      __builtin_prefetch(bSrc0 + (size_t)(kt + 2) * 32 * ldb, 0, 3);
    compute(cur);
    if (kt + 1 < NK) { storeA(1 - cur); storeB(1 - cur); }
    __syncthreads();
  }

  // fused clamped SwiGLU: gate = tiles 0..3, up = tiles 4..7
  #pragma unroll
  for (int j = 0; j < 4; ++j) {
    const int f = f0 + j * 16 + ln;
    #pragma unroll
    for (int r = 0; r < 8; ++r) {
      const int gRow = mtile * 128 + wave * 16 + r + 8 * h;
      if (gRow < rows) {
        float g = acc[j][r];
        float u = acc[j + 4][r];
        g = fminf(g, SWIGLU_LIMIT);
        u = fminf(fmaxf(u, -SWIGLU_LIMIT), SWIGLU_LIMIT);
        float sg  = 1.0f / (1.0f + __expf(-SWIGLU_ALPHA * g));
        float val = (u + 1.0f) * (g * sg);
        act[(size_t)(slotBase + gRow) * FDIM + f] = f2bf(val);
      }
    }
  }
}

// mlp2: out[tok, :] += wt * act[slot, :] @ w2[e]
// A tile (bf16 act) streamed global->LDS with async copies (ASYNCcnt).
__global__ void __launch_bounds__(256, 1)
moe_mlp2_kernel(const unsigned short* __restrict__ act,
                const float* __restrict__ w2,
                const int*   __restrict__ counts,
                const int*   __restrict__ offsets,
                const int*   __restrict__ tok_list,
                const float* __restrict__ wt_list,
                float* __restrict__ out) {
  const int e     = blockIdx.z;
  const int mtile = blockIdx.y;
  const int n0    = blockIdx.x * 128;
  const int rows  = counts[e];
  if (mtile * 128 >= rows) return;
  const int slotBase = offsets[e];

  __shared__ unsigned short ldsA[2][128 * ASTRIDE];
  __shared__ unsigned int   ldsB[2][2048];

  const int tid  = threadIdx.x;
  const int wave = tid >> 5;
  const int lane = tid & 31;
  const int h    = lane >> 4;
  const int ln   = lane & 15;

  const int arow = tid >> 1;
  const int acol = (tid & 1) * 16;
  const int gRowA = mtile * 128 + arow;
  const bool aValid = gRowA < rows;
  // invalid rows clamp to the expert's slot 0: their D rows are discarded below
  const unsigned short* aSrc =
      act + (size_t)(slotBase + (aValid ? gRowA : 0)) * FDIM + acol;

  const int kp = tid >> 4;
  const int nb = (tid & 15) * 8;
  const int bh = kp >> 3;
  const int bp = kp & 7;
  const size_t ldb = DDIM;
  const float* bSrc0 = w2 + (size_t)e * FDIM * ldb + (size_t)(2 * kp) * ldb + n0 + nb;

  v8f acc[8];
  #pragma unroll
  for (int j = 0; j < 8; ++j) acc[j] = (v8f){0,0,0,0,0,0,0,0};

  float4 bL0, bL1, bH0, bH1;

  // async copy of this thread's 32B of the A tile (2 x b128, IOFFSET hits both sides)
  auto asyncA = [&](int kt, int buf) {
    unsigned int la = lds_addr(&ldsA[buf][arow * ASTRIDE + acol]);
    unsigned long long ga = (unsigned long long)(uintptr_t)(aSrc + kt * 32);
    asm volatile("global_load_async_to_lds_b128 %0, %1, off\n\t"
                 "global_load_async_to_lds_b128 %0, %1, off offset:16"
                 :: "v"(la), "v"(ga) : "memory");
  };
  auto loadB = [&](int kt) {
    const float* r0 = bSrc0 + (size_t)kt * 32 * ldb;
    const float* r1 = r0 + ldb;
    bL0 = ((const float4*)r0)[0]; bL1 = ((const float4*)r0)[1];
    bH0 = ((const float4*)r1)[0]; bH1 = ((const float4*)r1)[1];
  };
  auto storeB = [&](int buf) {
    unsigned int* d = &ldsB[buf][0];
    const float lo[8] = {bL0.x,bL0.y,bL0.z,bL0.w, bL1.x,bL1.y,bL1.z,bL1.w};
    const float hi[8] = {bH0.x,bH0.y,bH0.z,bH0.w, bH1.x,bH1.y,bH1.z,bH1.w};
    #pragma unroll
    for (int i = 0; i < 8; ++i)
      d[((nb + i) * 2 + bh) * 8 + bp] = pk2(lo[i], hi[i]);
  };
  auto compute = [&](int buf) {
    FragBf a;
    #pragma unroll
    for (int v = 0; v < 8; ++v) {
      int kk = 16 * h + 2 * v;
      a.u[v] = *(const unsigned int*)&ldsA[buf][(wave * 16 + ln) * ASTRIDE + kk];
    }
    #pragma unroll
    for (int j = 0; j < 8; ++j) {
      FragBf b;
      const unsigned int* bpt = &ldsB[buf][((j * 16 + ln) * 2 + h) * 8];
      #pragma unroll
      for (int v = 0; v < 8; ++v) b.u[v] = bpt[v];
      acc[j] = __builtin_amdgcn_wmma_f32_16x16x32_bf16(
                 false, a.v, false, b.v, (short)0, acc[j], false, false);
    }
  };

  const int NK = FDIM / 32;
  asyncA(0, 0); loadB(0); storeB(0);
  asm volatile("s_wait_asynccnt 0" ::: "memory");
  __syncthreads();
  for (int kt = 0; kt < NK; ++kt) {
    const int cur = kt & 1;
    if (kt + 1 < NK) { asyncA(kt + 1, 1 - cur); loadB(kt + 1); }
    if (kt + 2 < NK)
      __builtin_prefetch(bSrc0 + (size_t)(kt + 2) * 32 * ldb, 0, 3);
    compute(cur);
    if (kt + 1 < NK) { storeB(1 - cur); }
    asm volatile("s_wait_asynccnt 0" ::: "memory");
    __syncthreads();
  }

  // weighted combine into output (each token touched by exactly top_k experts)
  #pragma unroll
  for (int j = 0; j < 8; ++j) {
    const int n = n0 + j * 16 + ln;
    #pragma unroll
    for (int r = 0; r < 8; ++r) {
      const int gRow = mtile * 128 + wave * 16 + r + 8 * h;
      if (gRow < rows) {
        const int slot = slotBase + gRow;
        const int tok  = tok_list[slot];
        const float wt = wt_list[slot];
        atomicAdd(&out[(size_t)tok * DDIM + n], acc[j][r] * wt);
      }
    }
  }
}

// ---------------------------------------------------------------- launch ----
extern "C" void kernel_launch(void* const* d_in, const int* in_sizes, int n_in,
                              void* d_out, int out_size, void* d_ws, size_t ws_size,
                              hipStream_t stream) {
  const float* x  = (const float*)d_in[0];
  const float* rw = (const float*)d_in[1];
  const float* rb = (const float*)d_in[2];
  const float* w1 = (const float*)d_in[3];
  const float* w2 = (const float*)d_in[4];
  const int T = in_sizes[0] / DDIM;
  float* out = (float*)d_out;

  char* ws = (char*)d_ws;
  size_t o = 0;
  int*   counts   = (int*)(ws + o);            o += 64;
  int*   cursors  = (int*)(ws + o);            o += 64;
  int*   offsets  = (int*)(ws + o);            o += 128;         // pad to 256
  int*   topk_idx = (int*)(ws + o);            o += (size_t)T * TOPK * 4;
  float* topk_wt  = (float*)(ws + o);          o += (size_t)T * TOPK * 4;
  int*   tok_list = (int*)(ws + o);            o += (size_t)T * TOPK * 4;
  float* wt_list  = (float*)(ws + o);          o += (size_t)T * TOPK * 4;
  o = (o + 255) & ~(size_t)255;
  unsigned short* actbuf = (unsigned short*)(ws + o);  // (T*TOPK) x FDIM bf16

  zero_ctrl_kernel<<<1, 64, 0, stream>>>(counts, cursors);
  router_kernel<<<(T + 255) / 256, 256, 0, stream>>>(x, rw, rb, counts,
                                                     topk_idx, topk_wt, T);
  scan_kernel<<<1, 32, 0, stream>>>(counts, offsets, cursors);
  scatter_kernel<<<(T + 255) / 256, 256, 0, stream>>>(topk_idx, topk_wt, offsets,
                                                      cursors, tok_list, wt_list, T);
  zero_out_kernel<<<(T * DDIM + 255) / 256, 256, 0, stream>>>(out, T * DDIM);

  dim3 g1(FDIM / 64, (T + 127) / 128, NEXP);
  moe_mlp1_kernel<<<g1, 256, 0, stream>>>(x, w1, counts, offsets, tok_list, actbuf);

  dim3 g2(DDIM / 128, (T + 127) / 128, NEXP);
  moe_mlp2_kernel<<<g2, 256, 0, stream>>>(actbuf, w2, counts, offsets,
                                          tok_list, wt_list, out);
}
